// EGNNetwork_64244120814371
// MI455X (gfx1250) — matrix operand order repaired and checked by reading, not compiled, not run on hardware
//
#include <hip/hip_runtime.h>
#include <hip/hip_bf16.h>

// ---------------------------------------------------------------------------
// EGNN forward for gfx1250 (MI455X).  Heavy MLPs run as bf16 WMMA
// (v_wmma_f32_16x16x32_bf16), 16 rows (edges/nodes) per wave32 tile.
// Scatters use fp32 global atomics (accumulators are L2-resident).
// SiLU/normalization use v_rcp_f32 instead of the IEEE divide expansion.
// ---------------------------------------------------------------------------

#define N_NODES  50000
#define N_EDGES  800000
#define NODE_F   32
#define EDGE_F   16
#define HID      192
#define OUTF     96
#define N_GRAPHS 64
#define POOLW    (OUTF + 3)   // 99

typedef __bf16 bf16_t;
typedef bf16_t v16bf __attribute__((ext_vector_type(16)));
typedef float  v8f   __attribute__((ext_vector_type(8)));
typedef unsigned int u32x8 __attribute__((ext_vector_type(8)));

static __device__ __forceinline__ unsigned short f2bf(float f) {
    union { float f; unsigned u; } c; c.f = f;
    unsigned u = c.u;
    u += 0x7fffu + ((u >> 16) & 1u);          // round-to-nearest-even
    return (unsigned short)(u >> 16);
}
static __device__ __forceinline__ bf16_t bfbits(unsigned short s) {
    return __builtin_bit_cast(bf16_t, s);
}
// fast reciprocal: single v_rcp_f32 (inputs already bf16-rounded; plenty of precision)
static __device__ __forceinline__ float frcp(float x) {
    return __builtin_amdgcn_rcpf(x);
}
static __device__ __forceinline__ float silu(float v) {
    return v * frcp(1.0f + __expf(-v));
}

// One 16x16 output tile: A[16 x 32*ktiles] (row-major bf16 in LDS, width TWA)
// times swizzled weights W.  B fragments are contiguous 32B per lane.
static __device__ __forceinline__ v8f wmma_rowtile(const unsigned short* A, int TWA,
                                                   const bf16_t* W, int ktiles,
                                                   int nt, int NT, int lane) {
    v8f acc = {0.f, 0.f, 0.f, 0.f, 0.f, 0.f, 0.f, 0.f};
    const int m  = lane & 15;
    const int hk = (lane >> 4) * 8;
    for (int kt = 0; kt < ktiles; ++kt) {
        const unsigned int* rp32 = (const unsigned int*)(A + m * TWA + kt * 32 + hk);
        u32x8 av;
        av[0] = rp32[0];  av[1] = rp32[1];  av[2] = rp32[2];  av[3] = rp32[3];
        av[4] = rp32[8];  av[5] = rp32[9];  av[6] = rp32[10]; av[7] = rp32[11];
        v16bf a = __builtin_bit_cast(v16bf, av);
        v16bf b = *(const v16bf*)(W + ((size_t)(kt * NT + nt) * 32 + lane) * 16);
        acc = __builtin_amdgcn_wmma_f32_16x16x32_bf16(false, a, false, b,
                                                      (short)0, acc, false, false);
    }
    return acc;
}

// bias + SiLU on the C/D fragment, store bf16 into LDS tile (width TW).
static __device__ __forceinline__ void store_act_silu(v8f acc, const float* bias, int nt,
                                                      unsigned short* T, int TW, int lane) {
    const int col = nt * 16 + (lane & 15);
    const float bv = bias[col];
    const int rb = (lane >> 4) * 8;
#pragma unroll
    for (int r = 0; r < 8; ++r) {
        float v = silu(acc[r] + bv);
        T[(rb + r) * TW + col] = f2bf(v);
    }
}

// ---------------------------------------------------------------------------
// fp32 -> bf16 weight conversion with zero-padding, written in the per-lane
// swizzled B-fragment order used by wmma_rowtile.
// dst[((kt*NT+nt)*32 + L)*16 + e] = src[k*srcN + n],  k = kt*32+(e&8)*2+(L>>4)*8+(e&7),
// n = nt*16+(L&15);  zero if k>=Krows or n>=srcN.
// ---------------------------------------------------------------------------
__global__ void convert_swizzle_kernel(const float* __restrict__ src, bf16_t* __restrict__ dst,
                                       int Krows, int srcN, int KPAD, int N) {
    const int NT = N / 16;
    int i = blockIdx.x * blockDim.x + threadIdx.x;
    int total = KPAD * N;
    if (i >= total) return;
    int e = i & 15;
    int L = (i >> 4) & 31;
    int t = i >> 9;
    int nt = t % NT, kt = t / NT;
    int k = kt * 32 + (e & 8) * 2 + (L >> 4) * 8 + (e & 7);
    int n = nt * 16 + (L & 15);
    float v = (k < Krows && n < srcN) ? src[k * srcN + n] : 0.0f;
    dst[i] = bfbits(f2bf(v));
}

__global__ void zero_kernel(float* __restrict__ p, int n) {
    int i = blockIdx.x * blockDim.x + threadIdx.x;
    if (i < n) p[i] = 0.0f;
}

// ---------------------------------------------------------------------------
// Edge kernel: 4 waves/block, 16 edges per wave.  800000 = 12500*64 exactly.
// ---------------------------------------------------------------------------
template <int IN>
__global__ __launch_bounds__(128) void edge_kernel(
    const float* __restrict__ feat, const float* __restrict__ coordIn,
    const float* __restrict__ edge_attr,
    const int* __restrict__ srcI, const int* __restrict__ dstI,
    const bf16_t* __restrict__ ew1w, const float* __restrict__ ew1b,
    const bf16_t* __restrict__ ew2w, const float* __restrict__ ew2b,
    const bf16_t* __restrict__ cw1w, const float* __restrict__ cw1b,
    const bf16_t* __restrict__ cw2w,
    float* __restrict__ h_neigh, float* __restrict__ x_sum, float* __restrict__ deg)
{
    constexpr int KF    = 2 * IN + EDGE_F + 1;          // 81 or 209
    constexpr int KFPAD = (KF + 31) & ~31;              // 96 or 224
    constexpr int KT1   = KFPAD / 32;                   // 3 or 7
    constexpr int TW0   = (KFPAD > HID) ? KFPAD : HID;  // 192 or 224
    constexpr int TW1   = HID;                          // 192
    constexpr int PADC  = KFPAD - KF;                   // 15

    __shared__ __attribute__((aligned(16))) unsigned short sh_t0[4][16 * TW0];
    __shared__ __attribute__((aligned(16))) unsigned short sh_t1[4][16 * TW1];
    __shared__ float sh_xd[4][16][3];
    __shared__ int   sh_si[4][16];
    __shared__ int   sh_di[4][16];
    __shared__ float sh_w[4][16];

    const int lane = threadIdx.x & 31;
    const int wid  = threadIdx.x >> 5;
    const int e0   = (blockIdx.x * 4 + wid) * 16;
    unsigned short* T0 = sh_t0[wid];
    unsigned short* T1 = sh_t1[wid];

    if (lane < 16) { sh_si[wid][lane] = srcI[e0 + lane]; sh_di[wid][lane] = dstI[e0 + lane]; }

    // --- build f = [feat[src], feat[dst], radial, edge_attr, 0-pad] (bf16) ---
    for (int idx = lane; idx < 16 * IN; idx += 32) {
        int e = idx / IN, c = idx - e * IN;
        T0[e * TW0 + c]      = f2bf(feat[sh_si[wid][e] * IN + c]);
        T0[e * TW0 + IN + c] = f2bf(feat[sh_di[wid][e] * IN + c]);
    }
    if (lane < 16) {
        int s = sh_si[wid][lane], d = sh_di[wid][lane];
        float dx = coordIn[s * 3 + 0] - coordIn[d * 3 + 0];
        float dy = coordIn[s * 3 + 1] - coordIn[d * 3 + 1];
        float dz = coordIn[s * 3 + 2] - coordIn[d * 3 + 2];
        float r  = dx * dx + dy * dy + dz * dz;
        float inv = frcp(sqrtf(r) + 1e-30f);
        sh_xd[wid][lane][0] = dx * inv;
        sh_xd[wid][lane][1] = dy * inv;
        sh_xd[wid][lane][2] = dz * inv;
        T0[lane * TW0 + 2 * IN] = f2bf(r);
    }
    for (int idx = lane; idx < 16 * EDGE_F; idx += 32) {
        int e = idx >> 4, c = idx & 15;
        T0[e * TW0 + 2 * IN + 1 + c] = f2bf(edge_attr[(e0 + e) * EDGE_F + c]);
    }
    for (int idx = lane; idx < 16 * PADC; idx += 32) {
        int e = idx / PADC, c = idx - e * PADC;
        T0[e * TW0 + KF + c] = 0;
    }

    // --- stage 1: h1 = silu(f @ ew1 + b) -> T1 ---
    for (int nt = 0; nt < 12; ++nt) {
        v8f acc = wmma_rowtile(T0, TW0, ew1w, KT1, nt, 12, lane);
        store_act_silu(acc, ew1b, nt, T1, TW1, lane);
    }
    // --- stage 2: msg = silu(h1 @ ew2 + b) -> T0, scatter to h_neigh ---
    for (int nt = 0; nt < 12; ++nt) {
        v8f acc = wmma_rowtile(T1, TW1, ew2w, 6, nt, 12, lane);
        const int col = nt * 16 + (lane & 15);
        const float bv = ew2b[col];
        const int rb = (lane >> 4) * 8;
#pragma unroll
        for (int r = 0; r < 8; ++r) {
            float v = silu(acc[r] + bv);
            T0[(rb + r) * TW0 + col] = f2bf(v);
            atomicAdd(&h_neigh[sh_di[wid][rb + r] * HID + col], v);
        }
    }
    // --- stage 3: h2 = silu(msg @ cw1 + b) -> T1 ---
    for (int nt = 0; nt < 12; ++nt) {
        v8f acc = wmma_rowtile(T0, TW0, cw1w, 6, nt, 12, lane);
        store_act_silu(acc, cw1b, nt, T1, TW1, lane);
    }
    // --- stage 4: w = h2 @ cw2 (col 0 of padded 192x16 B), coord scatter ---
    {
        v8f acc = wmma_rowtile(T1, TW1, cw2w, 6, 0, 1, lane);
        if ((lane & 15) == 0) {
            const int rb = (lane >> 4) * 8;
#pragma unroll
            for (int r = 0; r < 8; ++r) sh_w[wid][rb + r] = acc[r];
        }
        if (lane < 16) {
            float wv = sh_w[wid][lane];
            int d = sh_di[wid][lane];
            atomicAdd(&x_sum[d * 3 + 0], wv * sh_xd[wid][lane][0]);
            atomicAdd(&x_sum[d * 3 + 1], wv * sh_xd[wid][lane][1]);
            atomicAdd(&x_sum[d * 3 + 2], wv * sh_xd[wid][lane][2]);
            atomicAdd(&deg[d], 1.0f);
        }
    }
}

// ---------------------------------------------------------------------------
// Node kernel: 4 waves/block, 16 nodes per wave.  50000 = 3125 tiles.
// ---------------------------------------------------------------------------
template <int IN>
__global__ __launch_bounds__(128) void node_kernel(
    const float* __restrict__ feat, const float* __restrict__ coordIn,
    const float* __restrict__ h_neigh, const float* __restrict__ x_sum,
    const float* __restrict__ deg,
    const bf16_t* __restrict__ nw1w, const float* __restrict__ nw1b,
    const bf16_t* __restrict__ nw2w, const float* __restrict__ nw2b,
    float* __restrict__ featOut, float* __restrict__ coordOut)
{
    constexpr int K   = IN + HID;   // 224 or 288 (multiples of 32)
    constexpr int KT  = K / 32;
    constexpr int TW0 = K;
    constexpr int TW1 = HID;

    __shared__ __attribute__((aligned(16))) unsigned short sh_t0[4][16 * TW0];
    __shared__ __attribute__((aligned(16))) unsigned short sh_t1[4][16 * TW1];

    const int lane = threadIdx.x & 31;
    const int wid  = threadIdx.x >> 5;
    const int tile = blockIdx.x * 4 + wid;
    if (tile >= N_NODES / 16) return;
    const int n0 = tile * 16;
    unsigned short* T0 = sh_t0[wid];
    unsigned short* T1 = sh_t1[wid];

    for (int idx = lane; idx < 16 * IN; idx += 32) {
        int e = idx / IN, c = idx - e * IN;
        T0[e * TW0 + c] = f2bf(feat[(n0 + e) * IN + c]);
    }
    for (int idx = lane; idx < 16 * HID; idx += 32) {
        int e = idx / HID, c = idx - e * HID;
        T0[e * TW0 + IN + c] = f2bf(h_neigh[(n0 + e) * HID + c]);
    }

    for (int nt = 0; nt < 12; ++nt) {
        v8f acc = wmma_rowtile(T0, TW0, nw1w, KT, nt, 12, lane);
        store_act_silu(acc, nw1b, nt, T1, TW1, lane);
    }
    for (int nt = 0; nt < 6; ++nt) {
        v8f acc = wmma_rowtile(T1, TW1, nw2w, 6, nt, 6, lane);
        const int col = nt * 16 + (lane & 15);
        const float bv = nw2b[col];
        const int rb = (lane >> 4) * 8;
#pragma unroll
        for (int r = 0; r < 8; ++r) {
            float v = acc[r] + bv;
            featOut[(n0 + rb + r) * OUTF + col] = fmaxf(v, 0.0f);   // relu(feat)
        }
    }
    for (int idx = lane; idx < 48; idx += 32) {
        int e = idx / 3, j = idx - e * 3;
        int n = n0 + e;
        float dg = fmaxf(deg[n], 1.0f);
        coordOut[n * 3 + j] = coordIn[n * 3 + j] + x_sum[n * 3 + j] * frcp(dg);
    }
}

// ---------------------------------------------------------------------------
// Head: global_add_pool then tiny MLP.
// ---------------------------------------------------------------------------
__global__ void pool_kernel(const float* __restrict__ feat, const float* __restrict__ coord,
                            const int* __restrict__ batch, float* __restrict__ pooled) {
    int i = blockIdx.x * blockDim.x + threadIdx.x;
    if (i >= N_NODES * POOLW) return;
    int n = i / POOLW, c = i - n * POOLW;
    float v = (c < OUTF) ? feat[n * OUTF + c] : coord[n * 3 + (c - OUTF)];
    atomicAdd(&pooled[batch[n] * POOLW + c], v);
}

__global__ void head_kernel(const float* __restrict__ pooled,
                            const float* __restrict__ w1, const float* __restrict__ b1,
                            const float* __restrict__ w2, const float* __restrict__ b2,
                            float* __restrict__ out) {
    int g = threadIdx.x;
    if (g >= N_GRAPHS) return;
    float o = b2[0];
    for (int j = 0; j < 64; ++j) {
        float a = b1[j];
        for (int k = 0; k < POOLW; ++k) a += pooled[g * POOLW + k] * w1[k * 64 + j];
        o += fmaxf(a, 0.0f) * w2[j];
    }
    out[g] = o;
}

// ---------------------------------------------------------------------------
// Host side
// ---------------------------------------------------------------------------
extern "C" void kernel_launch(void* const* d_in, const int* in_sizes, int n_in,
                              void* d_out, int out_size, void* d_ws, size_t ws_size,
                              hipStream_t stream) {
    (void)in_sizes; (void)n_in; (void)out_size; (void)ws_size;

    const float* feat      = (const float*)d_in[0];
    const float* coord     = (const float*)d_in[1];
    const float* edge_attr = (const float*)d_in[2];
    const int*   eidx      = (const int*)d_in[3];
    const int*   batch     = (const int*)d_in[4];

    // params flattened by sorted dict keys: head{w1(w,b),w2(w,b)},
    // layers[i]{cw1(w,b), cw2(w), ew1(w,b), ew2(w,b), nw1(w,b), nw2(w,b)}
    int p = 5;
    const float* hw1 = (const float*)d_in[p++];
    const float* hb1 = (const float*)d_in[p++];
    const float* hw2 = (const float*)d_in[p++];
    const float* hb2 = (const float*)d_in[p++];
    struct LP { const float *cw1w,*cw1b,*cw2w,*ew1w,*ew1b,*ew2w,*ew2b,*nw1w,*nw1b,*nw2w,*nw2b; } L[3];
    for (int i = 0; i < 3; ++i) {
        L[i].cw1w = (const float*)d_in[p++]; L[i].cw1b = (const float*)d_in[p++];
        L[i].cw2w = (const float*)d_in[p++];
        L[i].ew1w = (const float*)d_in[p++]; L[i].ew1b = (const float*)d_in[p++];
        L[i].ew2w = (const float*)d_in[p++]; L[i].ew2b = (const float*)d_in[p++];
        L[i].nw1w = (const float*)d_in[p++]; L[i].nw1b = (const float*)d_in[p++];
        L[i].nw2w = (const float*)d_in[p++]; L[i].nw2b = (const float*)d_in[p++];
    }

    // workspace carve-up
    char* base = (char*)d_ws;
    size_t off = 0;
    auto alloc = [&](size_t bytes) -> void* {
        off = (off + 255) & ~(size_t)255;
        void* q = base + off;
        off += bytes;
        return q;
    };
    float* featA   = (float*)alloc((size_t)N_NODES * OUTF * 4);
    float* featB   = (float*)alloc((size_t)N_NODES * OUTF * 4);
    float* coordA  = (float*)alloc((size_t)N_NODES * 3 * 4);
    float* coordB  = (float*)alloc((size_t)N_NODES * 3 * 4);
    float* h_neigh = (float*)alloc((size_t)N_NODES * HID * 4);
    float* x_sum   = (float*)alloc((size_t)N_NODES * 3 * 4);
    float* degv    = (float*)alloc((size_t)N_NODES * 4);
    float* pooled  = (float*)alloc((size_t)N_GRAPHS * POOLW * 4);

    const int KPADe1[3]  = {96, 224, 224};         // padded 2*IN+17
    const int Ke1[3]     = {81, 209, 209};
    const int Knw1[3]    = {NODE_F + HID, OUTF + HID, OUTF + HID}; // 224, 288, 288

    bf16_t *ew1w16[3], *ew2w16[3], *cw1w16[3], *cw2w16[3], *nw1w16[3], *nw2w16[3];
    for (int i = 0; i < 3; ++i) {
        ew1w16[i] = (bf16_t*)alloc((size_t)KPADe1[i] * HID * 2);
        ew2w16[i] = (bf16_t*)alloc((size_t)HID * HID * 2);
        cw1w16[i] = (bf16_t*)alloc((size_t)HID * HID * 2);
        cw2w16[i] = (bf16_t*)alloc((size_t)HID * 16 * 2);
        nw1w16[i] = (bf16_t*)alloc((size_t)Knw1[i] * HID * 2);
        nw2w16[i] = (bf16_t*)alloc((size_t)HID * OUTF * 2);
    }

    auto conv = [&](const float* s, bf16_t* d, int Krows, int srcN, int KPAD, int N) {
        int total = KPAD * N;
        convert_swizzle_kernel<<<(total + 255) / 256, 256, 0, stream>>>(s, d, Krows, srcN, KPAD, N);
    };
    for (int i = 0; i < 3; ++i) {
        conv(L[i].ew1w, ew1w16[i], Ke1[i], HID, KPADe1[i], HID);
        conv(L[i].ew2w, ew2w16[i], HID, HID, HID, HID);
        conv(L[i].cw1w, cw1w16[i], HID, HID, HID, HID);
        conv(L[i].cw2w, cw2w16[i], HID, 1, HID, 16);
        conv(L[i].nw1w, nw1w16[i], Knw1[i], HID, Knw1[i], HID);
        conv(L[i].nw2w, nw2w16[i], HID, OUTF, HID, OUTF);
    }

    auto zero = [&](float* q, int n) {
        zero_kernel<<<(n + 255) / 256, 256, 0, stream>>>(q, n);
    };

    const int* srcI = eidx;
    const int* dstI = eidx + N_EDGES;
    const float* curFeat  = feat;
    const float* curCoord = coord;

    const int edgeGrid = N_EDGES / 64;                 // 12500
    const int nodeGrid = (N_NODES / 16 + 3) / 4;       // 782

    for (int i = 0; i < 3; ++i) {
        zero(h_neigh, N_NODES * HID);
        zero(x_sum,   N_NODES * 3);
        zero(degv,    N_NODES);

        float* fOut = (i % 2 == 0) ? featA  : featB;
        float* cOut = (i % 2 == 0) ? coordA : coordB;

        if (i == 0) {
            edge_kernel<NODE_F><<<edgeGrid, 128, 0, stream>>>(
                curFeat, curCoord, edge_attr, srcI, dstI,
                ew1w16[i], L[i].ew1b, ew2w16[i], L[i].ew2b,
                cw1w16[i], L[i].cw1b, cw2w16[i],
                h_neigh, x_sum, degv);
            node_kernel<NODE_F><<<nodeGrid, 128, 0, stream>>>(
                curFeat, curCoord, h_neigh, x_sum, degv,
                nw1w16[i], L[i].nw1b, nw2w16[i], L[i].nw2b, fOut, cOut);
        } else {
            edge_kernel<OUTF><<<edgeGrid, 128, 0, stream>>>(
                curFeat, curCoord, edge_attr, srcI, dstI,
                ew1w16[i], L[i].ew1b, ew2w16[i], L[i].ew2b,
                cw1w16[i], L[i].cw1b, cw2w16[i],
                h_neigh, x_sum, degv);
            node_kernel<OUTF><<<nodeGrid, 128, 0, stream>>>(
                curFeat, curCoord, h_neigh, x_sum, degv,
                nw1w16[i], L[i].nw1b, nw2w16[i], L[i].nw2b, fOut, cOut);
        }
        curFeat  = fOut;
        curCoord = cOut;
    }

    zero(pooled, N_GRAPHS * POOLW);
    {
        int total = N_NODES * POOLW;
        pool_kernel<<<(total + 255) / 256, 256, 0, stream>>>(curFeat, curCoord, batch, pooled);
    }
    head_kernel<<<1, 64, 0, stream>>>(pooled, hw1, hb1, hw2, hb2, (float*)d_out);
}
